// ViolenceDetectionGNN_31190052504456
// MI455X (gfx1250) — matrix operand order repaired and checked by reading, not compiled, not run on hardware
//
#include <hip/hip_runtime.h>
#include <hip/hip_bf16.h>

typedef __attribute__((ext_vector_type(16))) _Float16 v16h;
typedef __attribute__((ext_vector_type(8)))  float    v8f;

#define NN  150000
#define NE  1250000
#define NG  4096
#define HID 64

// ---------------------------------------------------------------- utilities
__global__ void k_zero(float* __restrict__ p, int n) {
    int i = blockIdx.x * 256 + threadIdx.x;
    if (i < n) p[i] = 0.0f;
}

// deg[dst] += 1 per edge
__global__ void k_degree(const int* __restrict__ ei, float* __restrict__ deg) {
    int e = blockIdx.x * 256 + threadIdx.x;
    if (e < NE) atomicAdd(&deg[ei[NE + e]], 1.0f);
}

// dinv = rsqrt(deg + 1)  (in place)
__global__ void k_dinv(float* __restrict__ d) {
    int i = blockIdx.x * 256 + threadIdx.x;
    if (i < NN) d[i] = rsqrtf(d[i] + 1.0f);
}

// ------------------------------------------------- layer 1: h = x @ W1  (K=2)
__global__ void k_lin1(const float* __restrict__ x, const float* __restrict__ W1,
                       float* __restrict__ h) {
    int t = blockIdx.x * 256 + threadIdx.x;        // NN*16 threads
    int n = t >> 4, q = (t & 15) * 4;
    if (n >= NN) return;
    float x0 = x[2 * n], x1 = x[2 * n + 1];
    float4 w0 = *(const float4*)(W1 + q);
    float4 w1 = *(const float4*)(W1 + 64 + q);
    float4 o;
    o.x = x0 * w0.x + x1 * w1.x;
    o.y = x0 * w0.y + x1 * w1.y;
    o.z = x0 * w0.z + x1 * w1.z;
    o.w = x0 * w0.w + x1 * w1.w;
    *(float4*)(h + n * 64 + q) = o;
}

// -------------------------------------- agg = h * dinv^2 + b  (self loop + bias)
__global__ void k_init_agg(const float* __restrict__ h, const float* __restrict__ dinv,
                           const float* __restrict__ b, float* __restrict__ agg) {
    int t = blockIdx.x * 256 + threadIdx.x;        // NN*16 threads
    int n = t >> 4, q = (t & 15) * 4;
    if (n >= NN) return;
    float s = dinv[n]; s *= s;
    float4 hv = *(const float4*)(h + n * 64 + q);
    float4 bv = *(const float4*)(b + q);
    float4 o = { hv.x * s + bv.x, hv.y * s + bv.y, hv.z * s + bv.z, hv.w * s + bv.w };
    *(float4*)(agg + n * 64 + q) = o;
}

// -------------------------------------- scatter: agg[dst] += h[src] * coef
__global__ void k_scatter(const int* __restrict__ ei, const float* __restrict__ h,
                          const float* __restrict__ dinv, float* __restrict__ agg) {
    int t = blockIdx.x * 256 + threadIdx.x;        // NE*16 threads
    int e = t >> 4, q = t & 15;
    if (e >= NE) return;
    int s = ei[e], d = ei[NE + e];
    float coef = dinv[s] * dinv[d];
    float4 hv = *(const float4*)(h + (long)s * 64 + q * 4);
    float* base = agg + (long)d * 64 + q * 4;
    atomicAdd(base + 0, hv.x * coef);
    atomicAdd(base + 1, hv.y * coef);
    atomicAdd(base + 2, hv.z * coef);
    atomicAdd(base + 3, hv.w * coef);
}

// ---------------------------------------------------------------- WMMA GEMM
// h = relu(A) @ W   where A = [NN x 64] f32, W = [64 x 64] f32 (row-major k,n).
// One wave -> 16 rows x 64 cols; 8 waves per block -> 128 rows per block.
// W is staged into LDS pre-swizzled into B-fragment layout so each wave reads
// each fragment as one aligned 32-byte v16h (2x ds_load_b128) instead of 16
// scalar ds_load_u16.
__global__ __launch_bounds__(256) void k_gemm_relu(const float* __restrict__ A,
                                                   const float* __restrict__ W,
                                                   float* __restrict__ H) {
    // 8 fragments x 32 lanes x 16 halves = 8 KB
    __shared__ _Float16 sWf[8 * 32 * 16];
    int tid  = threadIdx.x;
    int wave = tid >> 5, lane = tid & 31;
    int hi = (lane >= 16) ? 1 : 0;                 // lane half
    int r  = lane & 15;

    // Stage: wave w builds fragment f = w (identical for all waves at use time).
    // Fragment f: kc = f>>2 (K chunk of 32), nt = f&3 (N tile of 16).
    // B layout (32x16 f16): lane holds column r; half-slot i -> K = kc*32 + hi*16 + i.
    {
        int kc  = wave >> 2, nt = wave & 3;
        int col = nt * 16 + r;
        int kb  = kc * 32 + hi * 16;
        _Float16* dst = sWf + ((size_t)tid * 16);
#pragma unroll
        for (int i = 0; i < 16; ++i) dst[i] = (_Float16)W[(kb + i) * 64 + col];
    }
    __syncthreads();

    int m0 = blockIdx.x * 128 + wave * 16;
    if (m0 >= NN) return;                          // wave-uniform guard

    // Vector-load B fragments from swizzled LDS (32B aligned per lane).
    v16h Bf[2][4];
#pragma unroll
    for (int kc = 0; kc < 2; ++kc)
#pragma unroll
        for (int nt = 0; nt < 4; ++nt)
            Bf[kc][nt] = *(const v16h*)(sWf + (((kc * 4 + nt) * 32 + lane) * 16));

    // A fragments (16x32, f16) with fused ReLU: lane holds row m0+r;
    // halves 0..15 -> K = kc*32 + 16*hi + i
    const float* arow = A + (long)(m0 + r) * 64;
    v16h Af[2];
#pragma unroll
    for (int kc = 0; kc < 2; ++kc) {
        int off = kc * 32 + hi * 16;
        v16h a;
#pragma unroll
        for (int i = 0; i < 8; ++i) a[i]     = (_Float16)fmaxf(arow[off + i], 0.0f);
#pragma unroll
        for (int i = 0; i < 8; ++i) a[8 + i] = (_Float16)fmaxf(arow[off + 8 + i], 0.0f);
        Af[kc] = a;
    }

#pragma unroll
    for (int nt = 0; nt < 4; ++nt) {
        v8f c = {};
        c = __builtin_amdgcn_wmma_f32_16x16x32_f16(false, Af[0], false, Bf[0][nt],
                                                   (short)0, c, false, false);
        c = __builtin_amdgcn_wmma_f32_16x16x32_f16(false, Af[1], false, Bf[1][nt],
                                                   (short)0, c, false, false);
        // C/D layout: lane col = r; VGPR j -> row m0 + j + hi*8
#pragma unroll
        for (int j = 0; j < 8; ++j)
            H[(long)(m0 + j + hi * 8) * 64 + nt * 16 + r] = c[j];
    }
}

// -------------------------------------- pool: pooled[batch[n]] += relu(agg[n])
__global__ void k_pool(const int* __restrict__ batch, const float* __restrict__ agg,
                       float* __restrict__ pooled, float* __restrict__ cnts) {
    int t = blockIdx.x * 256 + threadIdx.x;        // NN*16 threads
    int n = t >> 4, q = (t & 15) * 4;
    if (n >= NN) return;
    int g = batch[n];
    float4 v = *(const float4*)(agg + (long)n * 64 + q);
    float* base = pooled + (long)g * 64 + q;
    atomicAdd(base + 0, fmaxf(v.x, 0.0f));
    atomicAdd(base + 1, fmaxf(v.y, 0.0f));
    atomicAdd(base + 2, fmaxf(v.z, 0.0f));
    atomicAdd(base + 3, fmaxf(v.w, 0.0f));
    if (q == 0) atomicAdd(&cnts[g], 1.0f);
}

// -------------------------------------- MLP head: sigmoid(relu(p@lw1+lb1)@lw2+lb2)
__global__ void k_mlp(const float* __restrict__ pooled, const float* __restrict__ cnts,
                      const float* __restrict__ lw1, const float* __restrict__ lb1,
                      const float* __restrict__ lw2, const float* __restrict__ lb2,
                      float* __restrict__ out) {
    int g = blockIdx.x * 256 + threadIdx.x;
    if (g >= NG) return;
    float inv = 1.0f / fmaxf(cnts[g], 1.0f);
    float p[64];
#pragma unroll
    for (int k = 0; k < 64; ++k) p[k] = pooled[(long)g * 64 + k] * inv;
    float o = lb2[0];
    for (int j = 0; j < 32; ++j) {
        float a = lb1[j];
#pragma unroll
        for (int k = 0; k < 64; ++k) a += p[k] * lw1[k * 32 + j];
        o += fmaxf(a, 0.0f) * lw2[j];
    }
    out[g] = 1.0f / (1.0f + expf(-o));
}

// ---------------------------------------------------------------- launcher
extern "C" void kernel_launch(void* const* d_in, const int* in_sizes, int n_in,
                              void* d_out, int out_size, void* d_ws, size_t ws_size,
                              hipStream_t stream) {
    const float* x    = (const float*)d_in[0];
    const int*   ei   = (const int*)d_in[1];
    const int*   bat  = (const int*)d_in[2];
    const float* W1   = (const float*)d_in[3];
    const float* b1   = (const float*)d_in[4];
    const float* W2   = (const float*)d_in[5];
    const float* b2   = (const float*)d_in[6];
    const float* W3   = (const float*)d_in[7];
    const float* b3   = (const float*)d_in[8];
    const float* lw1  = (const float*)d_in[9];
    const float* lb1  = (const float*)d_in[10];
    const float* lw2  = (const float*)d_in[11];
    const float* lb2  = (const float*)d_in[12];
    float* out = (float*)d_out;

    float* wsf    = (float*)d_ws;
    float* dinv   = wsf;                       // NN  (deg -> dinv in place)
    float* h      = wsf + 150016;              // NN*64
    float* agg    = h   + (size_t)NN * 64;     // NN*64
    float* pooled = agg + (size_t)NN * 64;     // NG*64
    float* cnts   = pooled + (size_t)NG * 64;  // NG

    const int T16   = NN * 16;                 // node x 16-lane kernels
    const int gN16  = (T16 + 255) / 256;       // 9375
    const int gE16  = (NE * 16 + 255) / 256;   // 78125
    const int gGemm = (NN + 127) / 128;        // 1172

    // degree / norm
    k_zero<<<(NN + 255) / 256, 256, 0, stream>>>(dinv, NN);
    k_zero<<<(NG * 64 + 255) / 256, 256, 0, stream>>>(pooled, NG * 64);
    k_zero<<<(NG + 255) / 256, 256, 0, stream>>>(cnts, NG);
    k_degree<<<(NE + 255) / 256, 256, 0, stream>>>(ei, dinv);
    k_dinv<<<(NN + 255) / 256, 256, 0, stream>>>(dinv);

    // layer 1 (K=2, VALU GEMM)
    k_lin1<<<gN16, 256, 0, stream>>>(x, W1, h);
    k_init_agg<<<gN16, 256, 0, stream>>>(h, dinv, b1, agg);
    k_scatter<<<gE16, 256, 0, stream>>>(ei, h, dinv, agg);

    // layer 2 (WMMA GEMM, ReLU fused on A-load)
    k_gemm_relu<<<gGemm, 256, 0, stream>>>(agg, W2, h);
    k_init_agg<<<gN16, 256, 0, stream>>>(h, dinv, b2, agg);
    k_scatter<<<gE16, 256, 0, stream>>>(ei, h, dinv, agg);

    // layer 3
    k_gemm_relu<<<gGemm, 256, 0, stream>>>(agg, W3, h);
    k_init_agg<<<gN16, 256, 0, stream>>>(h, dinv, b3, agg);
    k_scatter<<<gE16, 256, 0, stream>>>(ei, h, dinv, agg);

    // pool (ReLU fused) + MLP head
    k_pool<<<gN16, 256, 0, stream>>>(bat, agg, pooled, cnts);
    k_mlp<<<(NG + 255) / 256, 256, 0, stream>>>(pooled, cnts, lw1, lb1, lw2, lb2, out);
}
